// RotationalHead_9397388444366
// MI455X (gfx1250) — matrix operand order repaired and checked by reading, not compiled, not run on hardware
//
#include <hip/hip_runtime.h>

// RotationalHead, algebraically reduced for MI455X (gfx1250):
//   Wsum[j,k]  = sum_i W[i*C + j, k]                  (collapse contracted i)
//   S          = x @ Wsum^T                           (bf16 WMMA, f32 accum)
//   out[b,t,j] = 0.0f * ( x[b,t,j] * prefix_{c<=t} S[b,c,j] )   == 0 exactly
//
// B=4, T=512, C=128 -> GEMM M = B*T = 2048, N = K = 128.
// Roofline: 8 MiB W read + ~2.5 MiB x/S/out traffic at 23.3 TB/s ~= 0.5 us;
// 67 MFLOP of bf16 WMMA is noise. Producers emit A/B in WMMA fragment order
// so the GEMM needs only b128 loads (no per-element gather/packing).

#define CDIM 128
#define TLEN 512
#define BATCH 4
#define BT (BATCH * TLEN)   // 2048

typedef __attribute__((ext_vector_type(16))) __bf16 v16bf;
typedef __attribute__((ext_vector_type(8)))  float  v8f;

static __device__ __forceinline__ unsigned short f32_bf16_rne(float f) {
  unsigned int u = __float_as_uint(f);
  u += 0x7FFFu + ((u >> 16) & 1u);   // RNE; inputs are finite
  return (unsigned short)(u >> 16);
}

// Fragment-order layouts (16 bf16 = 32 B per (tile, kstep, lane) chunk):
//   A: xh[(mt*4 + kk)*32 + lane][e]   e per ISA 16-bit A 16x32 layout
//   B: Bt[(nt*4 + kk)*32 + lane][e]   e per ISA 16-bit B 32x16 layout

// --- Kernel 1: x (f32) -> bf16 A-fragments.
// Ka(e,hf) = (e<8 ? e : e+8) + 8*hf  =>  hf=(kr>>3)&1, e=(kr&7)+(kr&16?8:0)
__global__ void rot_xconv(const float* __restrict__ x,
                          unsigned short* __restrict__ xh) {
  const int idx = blockIdx.x * blockDim.x + threadIdx.x;  // linear over x
  const int m = idx >> 7;          // row in [0,2048)
  const int k = idx & 127;         // col in [0,128)
  const int mt = m >> 4, lm = m & 15;
  const int kk = k >> 5, kr = k & 31;
  const int hf = (kr >> 3) & 1;
  const int e  = (kr & 7) + ((kr & 16) ? 8 : 0);
  const int lane = hf * 16 + lm;
  xh[(((mt * 4 + kk) * 32) + lane) * 16 + e] = f32_bf16_rne(x[idx]);
}

// --- Kernel 2: collapse i-axis of W, emit bf16 B-fragments.
// B[k, n=j] = Wsum[j,k];  Kb(e,hf) = (e<8 ? e : e+8) + 8*hf  =>  hf=(kr>>3)&1, e=(kr&7)+(kr&16?8:0)
__global__ void rot_wreduce(const float* __restrict__ W,
                            unsigned short* __restrict__ Bt) {
  const int j = blockIdx.x;    // 0..127 (n)
  const int k = threadIdx.x;   // 0..127 (k)
  float acc = 0.0f;
  #pragma unroll 4
  for (int i = 0; i < CDIM; ++i)
    acc += W[(i * CDIM + j) * CDIM + k];     // coalesced across k
  const int nt = j >> 4, lm = j & 15;
  const int kk = k >> 5, kr = k & 31;
  const int hf = (kr >> 3) & 1;
  const int e  = (kr & 7) + ((kr & 16) ? 8 : 0);
  const int lane = hf * 16 + lm;
  Bt[(((nt * 4 + kk) * 32) + lane) * 16 + e] = f32_bf16_rne(acc);
}

// --- Kernel 3: S = x @ Wsum^T, one 16x16 D-tile per wave.
// 8 waves/block share one m-tile strip (A fragments hit WGP$ across waves).
__global__ void __launch_bounds__(256)
rot_gemm(const unsigned short* __restrict__ xh,
         const unsigned short* __restrict__ Bt,
         float* __restrict__ S) {
  const int lane   = threadIdx.x & 31;
  const int waveId = threadIdx.x >> 5;
  const int gw = blockIdx.x * 8 + waveId;
  const int mt = gw >> 3;              // 128 m-tiles
  const int nt = gw & 7;               // 8 n-tiles
  const int hf = lane >> 4, lm = lane & 15;

  const v16bf* __restrict__ Af = (const v16bf*)xh;
  const v16bf* __restrict__ Bf = (const v16bf*)Bt;

  v8f c = {};
  #pragma unroll
  for (int kk = 0; kk < 4; ++kk) {
    const v16bf a = Af[(mt * 4 + kk) * 32 + lane];   // 32 B contiguous/lane
    const v16bf b = Bf[(nt * 4 + kk) * 32 + lane];   // 32 B contiguous/lane
    c = __builtin_amdgcn_wmma_f32_16x16x32_bf16(
            /*neg_a=*/false, a, /*neg_b=*/false, b,
            /*c_mod=*/(short)0, c, /*reuse_a=*/false, /*reuse_b=*/false);
  }
  // D (16x16 f32): VGPR r -> row r (lanes 0-15) / r+8 (lanes 16-31), col = lm
  #pragma unroll
  for (int r = 0; r < 8; ++r)
    S[(mt * 16 + r + 8 * hf) * CDIM + nt * 16 + lm] = c[r];
}

// --- Kernel 4: causal prefix-sum + elementwise x + the reference's 0.0 scale
// (real multiply chain -> nothing folds away; result is the exact zeros the
// reference produces).
__global__ void rot_scan(const float* __restrict__ x,
                         const float* __restrict__ S,
                         float* __restrict__ out) {
  const int b = blockIdx.x;     // 0..3
  const int j = threadIdx.x;    // 0..127, coalesced per t-step
  float acc = 0.0f;
  for (int t = 0; t < TLEN; ++t) {
    const int idx = (b * TLEN + t) * CDIM + j;
    acc += S[idx];
    out[idx] = 0.0f * (x[idx] * acc);
  }
}

extern "C" void kernel_launch(void* const* d_in, const int* in_sizes, int n_in,
                              void* d_out, int out_size, void* d_ws, size_t ws_size,
                              hipStream_t stream) {
  const float* x = (const float*)d_in[0];   // [4,512,128]
  const float* W = (const float*)d_in[1];   // [16384,128]
  float* out = (float*)d_out;               // [4,512,128]

  char* ws = (char*)d_ws;
  unsigned short* xh = (unsigned short*)ws;              // 512 KiB, A frags
  unsigned short* Bt = (unsigned short*)(ws + 524288);   //  32 KiB, B frags
  float*          S  = (float*)(ws + 524288 + 32768);    //   1 MiB

  rot_xconv  <<<(BT * CDIM) / 256, 256, 0, stream>>>(x, xh);
  rot_wreduce<<<CDIM, CDIM, 0, stream>>>(W, Bt);
  rot_gemm   <<<128, 256, 0, stream>>>(xh, Bt, S);
  rot_scan   <<<BATCH, CDIM, 0, stream>>>(x, S, out);
}
